// BondDecoder_16003048144909
// MI455X (gfx1250) — matrix operand order repaired and checked by reading, not compiled, not run on hardware
//
#include <hip/hip_runtime.h>
#include <hip/hip_bf16.h>
#include <math.h>

// Problem constants (match reference)
#define LL 512
#define BB 16
#define DD 256
#define HH 4
#define HD 64
#define MAX_BONDS 6
#define NEG_BIG (-3.0e38f)

typedef __attribute__((ext_vector_type(16))) _Float16 v16h;
typedef __attribute__((ext_vector_type(8)))  float    v8f;

// CDNA5 WMMA 16x16x32 f16 fragment index maps (ISA 7.12.2):
// A (16x32, MxK): lane&15 = M (both halves), element i, group g=lane>>4:
//   K = 16*(i>>3) + 8*g + (i&7)
// B (32x16, KxN): N = lane&15, K = 16*(i>>3) + 8*g + (i&7)
// C (16x16 f32): row M = v + 8*g, col N = lane&15
__device__ __forceinline__ int a_kidx(int i, int g) { return ((i >> 3) << 4) + (g << 3) + (i & 7); }
__device__ __forceinline__ int b_kidx(int i, int g) { return ((i >> 3) << 4) + (g << 3) + (i & 7); }

__device__ __forceinline__ float rowmax16(float x) {
#pragma unroll
  for (int m = 1; m < 16; m <<= 1) x = fmaxf(x, __shfl_xor(x, m, 16));
  return x;
}
__device__ __forceinline__ float rowsum16(float x) {
#pragma unroll
  for (int m = 1; m < 16; m <<= 1) x += __shfl_xor(x, m, 16);
  return x;
}

// ---------------------------------------------------------------------------
// Kernel 1: fold the two consecutive linear layers:
//   Weff[o] = Wqk_slice(256x256) @ W(256x256), stored fp16.
// o: 0=Q_inc 1=K_inc 2=Q_dec 3=K_dec.  1 wave per 16x16 output tile.
// ---------------------------------------------------------------------------
__global__ __launch_bounds__(32) void fold_weights_kernel(
    const float* __restrict__ Wqk_inc, const float* __restrict__ Wq_inc,
    const float* __restrict__ Wk_inc,
    const float* __restrict__ Wqk_dec, const float* __restrict__ Wq_dec,
    const float* __restrict__ Wk_dec,
    _Float16* __restrict__ Weff /* [4][256][256] */) {
  const int blk = blockIdx.x;       // 0..1023
  const int o = blk >> 8;           // which fold
  const int t = blk & 255;          // tile id in 16x16 grid
  const int tm = (t >> 4) << 4;
  const int tn = (t & 15) << 4;

  const float* Aw;
  const float* Bw;
  int aoff;
  if (o == 0)      { Aw = Wqk_inc; aoff = 0;   Bw = Wq_inc; }
  else if (o == 1) { Aw = Wqk_inc; aoff = 256; Bw = Wk_inc; }
  else if (o == 2) { Aw = Wqk_dec; aoff = 0;   Bw = Wq_dec; }
  else             { Aw = Wqk_dec; aoff = 256; Bw = Wk_dec; }

  const int lane = threadIdx.x & 31;
  const int g = lane >> 4, n = lane & 15;

  v8f c = {};
  for (int k0 = 0; k0 < 256; k0 += 32) {
    v16h a, b;
#pragma unroll
    for (int i = 0; i < 16; ++i) {
      a[i] = (_Float16)Aw[(size_t)(tm + n) * 512 + aoff + k0 + a_kidx(i, g)];
      b[i] = (_Float16)Bw[(size_t)(k0 + b_kidx(i, g)) * 256 + tn + n];
    }
    c = __builtin_amdgcn_wmma_f32_16x16x32_f16(false, a, false, b, (short)0, c,
                                               false, false);
  }
#pragma unroll
  for (int v = 0; v < 8; ++v) {
    const int row = tm + v + 8 * g;
    Weff[(size_t)o * 65536 + (size_t)row * 256 + tn + n] = (_Float16)c[v];
  }
}

// ---------------------------------------------------------------------------
// Kernel 2: Y[o] = x @ Weff[o] + bias[o], fp16 out.  M=8192 (b*512+l), N=K=256.
// Block = 8 waves; each wave does 2 N-tiles of one M-tile.  grid = 512*4.
// x rows come from molecule_embedding[l, b, d] (L,B,D layout).
// ---------------------------------------------------------------------------
__global__ __launch_bounds__(256) void project_kernel(
    const float* __restrict__ emb, const _Float16* __restrict__ Weff,
    const float* __restrict__ bq_inc, const float* __restrict__ bk_inc,
    const float* __restrict__ bq_dec, const float* __restrict__ bk_dec,
    _Float16* __restrict__ Y /* [4][8192][256] */) {
  const int o = blockIdx.x & 3;
  const int m0 = (blockIdx.x >> 2) << 4;  // global row tile
  const float* bias = (o == 0) ? bq_inc : (o == 1) ? bk_inc
                       : (o == 2) ? bq_dec : bk_dec;
  const _Float16* W = Weff + (size_t)o * 65536;
  _Float16* Yo = Y + (size_t)o * 8192 * 256;

  const int wave = threadIdx.x >> 5;
  const int lane = threadIdx.x & 31;
  const int g = lane >> 4, n = lane & 15;

  // A row: global GEMM row m -> emb[(l*16 + b)*256], b=m>>9, l=m&511
  const int m = m0 + n;
  const float* arow = emb + ((size_t)(m & 511) * 16 + (size_t)(m >> 9)) * 256;

  // Preload A fragments once (reused for both N tiles)
  v16h afr[8];
#pragma unroll
  for (int ks = 0; ks < 8; ++ks)
#pragma unroll
    for (int i = 0; i < 16; ++i)
      afr[ks][i] = (_Float16)arow[ks * 32 + a_kidx(i, g)];

  for (int nt2 = 0; nt2 < 2; ++nt2) {
    const int n0 = ((wave << 1) + nt2) << 4;
    v8f c = {};
#pragma unroll
    for (int ks = 0; ks < 8; ++ks) {
      v16h b;
#pragma unroll
      for (int i = 0; i < 16; ++i)
        b[i] = W[(size_t)(ks * 32 + b_kidx(i, g)) * 256 + n0 + n];
      c = __builtin_amdgcn_wmma_f32_16x16x32_f16(false, afr[ks], false, b,
                                                 (short)0, c, false, false);
    }
    const float bv = bias[n0 + n];
#pragma unroll
    for (int v = 0; v < 8; ++v) {
      const int row = m0 + v + 8 * g;
      Yo[(size_t)row * 256 + n0 + n] = (_Float16)(c[v] + bv);
    }
  }
}

// ---------------------------------------------------------------------------
// Kernel 3: fused attention scores -> softmax -> head-mix diff -> bond
// log-probs -> output.  Block = (batch b, 16-row l-tile), 8 waves, wave w
// handles path p=w>>2 (0=inc,1=dec) and head h=w&3.  Never materializes the
// [B,H,L,L] attention tensors (saves ~270 MB of HBM traffic).
// ---------------------------------------------------------------------------
__global__ __launch_bounds__(256) void attn_diff_kernel(
    const _Float16* __restrict__ Y, const int* __restrict__ src_bond,
    const unsigned char* __restrict__ src_mask, const float* __restrict__ Wc,
    const float* __restrict__ bc, float* __restrict__ out) {
  const int b = blockIdx.x >> 5;
  const int l0 = (blockIdx.x & 31) << 4;

  __shared__ float sWc[16];
  __shared__ float sbc[4];
  __shared__ float sP[8][16][16];  // per-wave normalized attention tile
  __shared__ int sBond[16][MAX_BONDS];
  __shared__ float sPadL[16];

  const int tid = threadIdx.x;
  const int wave = tid >> 5;
  const int lane = tid & 31;
  const int g = lane >> 4, n = lane & 15;
  const int path = wave >> 2;
  const int head = wave & 3;

  if (tid < 16) {
    sWc[tid] = Wc[tid];
    sPadL[tid] = src_mask[b * LL + l0 + tid] ? 0.f : 1.f;
#pragma unroll
    for (int j = 0; j < MAX_BONDS; ++j)
      sBond[tid][j] = src_bond[(b * LL + l0 + tid) * MAX_BONDS + j];
  }
  if (tid < 4) sbc[tid] = bc[tid];

  const _Float16* Q = Y + (size_t)(path * 2 + 0) * 8192 * 256;
  const _Float16* K = Y + (size_t)(path * 2 + 1) * 8192 * 256;

  // A fragments: Q rows l0..l0+15, cols head*64..head*64+63 (K = 64 = 2 slabs)
  const _Float16* qrow =
      Q + (size_t)(b * LL + l0 + n) * 256 + head * HD;
  v16h a0, a1;
#pragma unroll
  for (int i = 0; i < 16; ++i) {
    a0[i] = qrow[a_kidx(i, g)];
    a1[i] = qrow[32 + a_kidx(i, g)];
  }

  float rmax[8], rsum[8];
#pragma unroll
  for (int v = 0; v < 8; ++v) { rmax[v] = NEG_BIG; rsum[v] = 0.f; }

  // ---------------- Pass 1: online softmax statistics ----------------
  for (int mt = 0; mt < 32; ++mt) {
    const int m0 = mt << 4;
    const _Float16* krow =
        K + (size_t)(b * LL + m0 + n) * 256 + head * HD;
    if (mt + 1 < 32) __builtin_prefetch(krow + 16 * 256, 0, 0);  // next K tile
    v16h b0, b1;
#pragma unroll
    for (int i = 0; i < 16; ++i) {
      b0[i] = krow[b_kidx(i, g)];        // B[k][n] = K[m0+n][k]
      b1[i] = krow[32 + b_kidx(i, g)];
    }
    v8f c = {};
    c = __builtin_amdgcn_wmma_f32_16x16x32_f16(false, a0, false, b0, (short)0,
                                               c, false, false);
    c = __builtin_amdgcn_wmma_f32_16x16x32_f16(false, a1, false, b1, (short)0,
                                               c, false, false);
    const bool kmask = src_mask[b * LL + m0 + n] != 0;  // this lane's column
#pragma unroll
    for (int v = 0; v < 8; ++v) {
      float s = kmask ? NEG_BIG : c[v] * 0.125f;  // 1/sqrt(64)
      const float tmax = rowmax16(s);
      const float nmax = fmaxf(rmax[v], tmax);
      const float e = __expf(s - nmax);           // masked -> underflow to 0
      const float esum = rowsum16(e);
      rsum[v] = rsum[v] * __expf(rmax[v] - nmax) + esum;
      rmax[v] = nmax;
    }
  }
  float rinv[8];
#pragma unroll
  for (int v = 0; v < 8; ++v) rinv[v] = 1.f / rsum[v];

  __syncthreads();  // sWc/sbc/sBond/sPadL visible before combine phase

  const float LOG07 = __logf(0.7f + 1e-6f);
  const float LOG01 = __logf(0.1f + 1e-6f);
  const float LOG025 = __logf(0.25f + 1e-6f);

  // ---------------- Pass 2: recompute, normalize, combine, emit ----------
  for (int mt = 0; mt < 32; ++mt) {
    const int m0 = mt << 4;
    const _Float16* krow =
        K + (size_t)(b * LL + m0 + n) * 256 + head * HD;
    v16h b0, b1;
#pragma unroll
    for (int i = 0; i < 16; ++i) {
      b0[i] = krow[b_kidx(i, g)];
      b1[i] = krow[32 + b_kidx(i, g)];
    }
    v8f c = {};
    c = __builtin_amdgcn_wmma_f32_16x16x32_f16(false, a0, false, b0, (short)0,
                                               c, false, false);
    c = __builtin_amdgcn_wmma_f32_16x16x32_f16(false, a1, false, b1, (short)0,
                                               c, false, false);
    const bool kmask = src_mask[b * LL + m0 + n] != 0;
#pragma unroll
    for (int v = 0; v < 8; ++v) {
      const float s = c[v] * 0.125f;
      const float p = kmask ? 0.f : __expf(s - rmax[v]) * rinv[v];
      sP[wave][v + 8 * g][n] = p;
    }
    __syncthreads();

    // One output element (l,m) per thread, 4 channels each.
    const int li = tid >> 4, mi = tid & 15;
    const int mglob = m0 + mi;
    const int lglob = l0 + li;
    const float padm = src_mask[b * LL + mglob] ? 0.f : 1.f;
    const float pm2 = sPadL[li] * padm;
    int cnt = 0;
#pragma unroll
    for (int j = 0; j < MAX_BONDS; ++j) cnt += (sBond[li][j] == mglob);
    const float srcw = (lglob == mglob) ? 0.f : (float)cnt * pm2;
    const int nidx = (int)srcw;
    const float scale = 4.f * pm2;

    float d[4];
#pragma unroll
    for (int cc = 0; cc < 4; ++cc) {
      float acc = sbc[cc];
#pragma unroll
      for (int hh = 0; hh < 4; ++hh)
        acc += (sP[hh][li][mi] - sP[4 + hh][li][mi]) * sWc[hh * 4 + cc];
      const float diffv = acc * scale;
      const float logp = (nidx < 4) ? ((cc == nidx) ? LOG07 : LOG01) : LOG025;
      d[cc] = logp + diffv;
    }
    float4* op =
        (float4*)(out + (((size_t)b * LL + lglob) * LL + mglob) * 4);
    *op = make_float4(d[0], d[1], d[2], d[3]);
    __syncthreads();  // sP reused next iteration
  }
}

// ---------------------------------------------------------------------------
extern "C" void kernel_launch(void* const* d_in, const int* in_sizes, int n_in,
                              void* d_out, int out_size, void* d_ws,
                              size_t ws_size, hipStream_t stream) {
  const float* emb       = (const float*)d_in[0];   // [512,16,256] f32
  const int*   src_bond  = (const int*)d_in[1];     // [16,512,6] i32
  const unsigned char* src_mask = (const unsigned char*)d_in[2];  // [16,512] bool
  const float* W_inc_qk  = (const float*)d_in[3];
  const float* Wq_inc    = (const float*)d_in[4];
  const float* bq_inc    = (const float*)d_in[5];
  const float* Wk_inc    = (const float*)d_in[6];
  const float* bk_inc    = (const float*)d_in[7];
  const float* W_dec_qk  = (const float*)d_in[8];
  const float* Wq_dec    = (const float*)d_in[9];
  const float* bq_dec    = (const float*)d_in[10];
  const float* Wk_dec    = (const float*)d_in[11];
  const float* bk_dec    = (const float*)d_in[12];
  const float* Wc        = (const float*)d_in[13];
  const float* bc        = (const float*)d_in[14];
  float* out = (float*)d_out;

  // Workspace layout (fp16): Weff[4][256*256] then Y[4][8192*256]
  _Float16* Weff = (_Float16*)d_ws;
  _Float16* Y = Weff + (size_t)4 * 65536;

  fold_weights_kernel<<<1024, 32, 0, stream>>>(W_inc_qk, Wq_inc, Wk_inc,
                                               W_dec_qk, Wq_dec, Wk_dec, Weff);
  project_kernel<<<2048, 256, 0, stream>>>(emb, Weff, bq_inc, bk_inc, bq_dec,
                                           bk_dec, Y);
  attn_diff_kernel<<<512, 256, 0, stream>>>(Y, src_bond, src_mask, Wc, bc, out);
}
